// TransformerBlockRBF_61186104098902
// MI455X (gfx1250) — matrix operand order repaired and checked by reading, not compiled, not run on hardware
//
#include <hip/hip_runtime.h>

// ---------------------------------------------------------------------------
// Types for gfx1250 WMMA / TDM (wave32)
// ---------------------------------------------------------------------------
typedef __attribute__((ext_vector_type(16))) __bf16       bf16x16;
typedef __attribute__((ext_vector_type(8)))  __bf16       bf16x8;
typedef __attribute__((ext_vector_type(8)))  float        f32x8;
typedef __attribute__((ext_vector_type(4)))  unsigned int v4u;
typedef __attribute__((ext_vector_type(8)))  int          v8i;
typedef __attribute__((ext_vector_type(4)))  int          v4i;

#define BQ   4
#define LQ   1024
#define DQ   768
#define HQ   12
#define HDQ  64
#define ROWS 4096   // B*L

__device__ __forceinline__ f32x8 wmma_bf16(bf16x16 a, bf16x16 b, f32x8 c) {
    return __builtin_amdgcn_wmma_f32_16x16x32_bf16(false, a, false, b,
                                                   (short)0, c, false, false);
}

// A-matrix 16x32 bf16 fragment (ISA 7.12.2): lane m = l&15,
// koff = (l>=16)?8:0 ; elements K = koff+0..7 and koff+16..23.
__device__ __forceinline__ bf16x16 frag_a(const __bf16* base, int ld, int lane) {
    int m    = lane & 15;
    int koff = (lane & 16) ? 8 : 0;
    const __bf16* p = base + m * ld + koff;
    bf16x8 lo = *(const bf16x8*)(p);
    bf16x8 hi = *(const bf16x8*)(p + 16);
    return __builtin_shufflevector(lo, hi, 0,1,2,3,4,5,6,7,8,9,10,11,12,13,14,15);
}

// B-matrix 32x16 bf16 fragment read from Bt[n][k] (row = N, col = K):
// lane n = l&15 ; koff = (l>=16)?8:0 ; elements K = koff+0..7 and koff+16..23.
__device__ __forceinline__ bf16x16 frag_b(const __bf16* base, int ld, int lane) {
    int n  = lane & 15;
    int kh = (lane & 16) ? 8 : 0;
    const __bf16* p = base + n * ld + kh;
    bf16x8 lo = *(const bf16x8*)(p);
    bf16x8 hi = *(const bf16x8*)(p + 16);
    return __builtin_shufflevector(lo, hi, 0,1,2,3,4,5,6,7,8,9,10,11,12,13,14,15);
}

// ---------------------------------------------------------------------------
// Tensor Data Mover: async DMA of a 2D bf16 tile (tile_dim0=TK cols, padded
// rows in LDS) from global [tensor_dim1 x DQ] into LDS.  ISA ch.8 D# layout.
// Rows are TK=64 bf16 = 128B = 32 DWORDs -> pad_interval code 4; LDS row pad
// is 16B = 4 DWORDs -> pad_amount code 3 (giving LDA/LDB = 72 bf16 rows).
// ---------------------------------------------------------------------------
#define TM  128
#define TN  64
#define TK  64
#define LDA 72
#define LDB 72

__device__ __forceinline__ void tdm_load_tile(const __bf16* gsrc, unsigned lds_off,
                                              int tensor_dim1, int tile_dim1) {
    unsigned long long ga = (unsigned long long)(uintptr_t)gsrc;
    v4u g0;
    g0.x = 1u;                                     // count=1, user descriptor
    g0.y = lds_off;                                // lds_addr (bytes)
    g0.z = (unsigned)ga;                           // global_addr[31:0]
    g0.w = (unsigned)((ga >> 32) & 0x01FFFFFFu)    // global_addr[56:32]
         | (2u << 30);                             // type = 2 ("image")
    v8i g1;
    g1[0] = (1 << 16)                              // data_size: 2 bytes
          | (1 << 20)                              // pad_enable
          | (4 << 22)                              // pad_interval: 32 DWORDs
          | (3 << 25);                             // pad_amount: 4 DWORDs
    g1[1] = (DQ & 0xFFFF) << 16;                   // tensor_dim0[15:0]
    g1[2] = (DQ >> 16) | ((tensor_dim1 & 0xFFFF) << 16);
    g1[3] = (tensor_dim1 >> 16) | (TK << 16);      // tile_dim0 = TK
    g1[4] = tile_dim1;                             // tile_dim1 ; tile_dim2 = 0
    g1[5] = DQ;                                    // tensor_dim0_stride[31:0]
    g1[6] = 0;
    g1[7] = 0;
    v4i gz4 = {0, 0, 0, 0};                        // 2D: groups 2/3 unused
    v8i gz8 = {0, 0, 0, 0, 0, 0, 0, 0};
    __builtin_amdgcn_tensor_load_to_lds(g0, g1, gz4, gz4, gz8, 0);
}

// ---------------------------------------------------------------------------
// Weight transpose + cast:  W[K=768][N=768] f32  ->  Wt[N][K] bf16
// ---------------------------------------------------------------------------
__global__ __launch_bounds__(256)
void k_transpose_bf16(const float* __restrict__ W, __bf16* __restrict__ Wt) {
    __shared__ float tile[32][33];
    int tx = threadIdx.x, ty = threadIdx.y;            // (32,8)
    int n  = blockIdx.x * 32 + tx;
    int k0 = blockIdx.y * 32;
    #pragma unroll
    for (int i = 0; i < 32; i += 8)
        tile[ty + i][tx] = W[(size_t)(k0 + ty + i) * DQ + n];
    __syncthreads();
    int n0 = blockIdx.x * 32;
    int k  = k0 + tx;
    #pragma unroll
    for (int i = 0; i < 32; i += 8)
        Wt[(size_t)(n0 + ty + i) * DQ + k] = (__bf16)tile[tx][ty + i];
}

// ---------------------------------------------------------------------------
// cf = centers / lengthscale  (bf16, natural [center][d] layout == Bt layout)
// csum[n] = sum_d cf[n][d]^2
// ---------------------------------------------------------------------------
__global__ __launch_bounds__(256)
void k_centers(const float* __restrict__ cen, const float* __restrict__ ls,
               __bf16* __restrict__ cfb, float* __restrict__ csum) {
    __shared__ float red[256];
    int n = blockIdx.x, t = threadIdx.x;
    float s = 0.f;
    #pragma unroll
    for (int i = 0; i < 3; ++i) {
        int col = t + i * 256;
        float c = cen[(size_t)n * DQ + col] / ls[col];
        cfb[(size_t)n * DQ + col] = (__bf16)c;
        s += c * c;
    }
    red[t] = s; __syncthreads();
    for (int off = 128; off > 0; off >>= 1) {
        if (t < off) red[t] += red[t + off];
        __syncthreads();
    }
    if (t == 0) csum[n] = red[0];
}

// ---------------------------------------------------------------------------
// LayerNorm 1: x (f32) -> bf16
// ---------------------------------------------------------------------------
__global__ __launch_bounds__(256)
void k_ln1(const float* __restrict__ x, const float* __restrict__ g,
           const float* __restrict__ b, __bf16* __restrict__ out) {
    __shared__ float rs[256], rs2[256];
    int row = blockIdx.x, t = threadIdx.x;
    const float* xr = x + (size_t)row * DQ;
    float v0 = xr[t], v1 = xr[t + 256], v2 = xr[t + 512];
    rs[t]  = v0 + v1 + v2;
    rs2[t] = v0 * v0 + v1 * v1 + v2 * v2;
    __syncthreads();
    for (int off = 128; off > 0; off >>= 1) {
        if (t < off) { rs[t] += rs[t + off]; rs2[t] += rs2[t + off]; }
        __syncthreads();
    }
    float mean = rs[0] * (1.f / DQ);
    float var  = rs2[0] * (1.f / DQ) - mean * mean;
    float inv  = rsqrtf(var + 1e-5f);
    __bf16* o = out + (size_t)row * DQ;
    o[t]       = (__bf16)(g[t]       * (v0 - mean) * inv + b[t]);
    o[t + 256] = (__bf16)(g[t + 256] * (v1 - mean) * inv + b[t + 256]);
    o[t + 512] = (__bf16)(g[t + 512] * (v2 - mean) * inv + b[t + 512]);
}

// ---------------------------------------------------------------------------
// LayerNorm 2 + ARD scaling: x2 -> hf bf16, rowsq[row] = sum(hf^2)
// ---------------------------------------------------------------------------
__global__ __launch_bounds__(256)
void k_ln2(const float* __restrict__ x, const float* __restrict__ g,
           const float* __restrict__ b, const float* __restrict__ ls,
           __bf16* __restrict__ out, float* __restrict__ rowsq) {
    __shared__ float rs[256], rs2[256];
    int row = blockIdx.x, t = threadIdx.x;
    const float* xr = x + (size_t)row * DQ;
    float v0 = xr[t], v1 = xr[t + 256], v2 = xr[t + 512];
    rs[t]  = v0 + v1 + v2;
    rs2[t] = v0 * v0 + v1 * v1 + v2 * v2;
    __syncthreads();
    for (int off = 128; off > 0; off >>= 1) {
        if (t < off) { rs[t] += rs[t + off]; rs2[t] += rs2[t + off]; }
        __syncthreads();
    }
    float mean = rs[0] * (1.f / DQ);
    float var  = rs2[0] * (1.f / DQ) - mean * mean;
    float inv  = rsqrtf(var + 1e-5f);
    float h0 = (g[t]       * (v0 - mean) * inv + b[t])       / ls[t];
    float h1 = (g[t + 256] * (v1 - mean) * inv + b[t + 256]) / ls[t + 256];
    float h2 = (g[t + 512] * (v2 - mean) * inv + b[t + 512]) / ls[t + 512];
    __bf16* o = out + (size_t)row * DQ;
    o[t] = (__bf16)h0; o[t + 256] = (__bf16)h1; o[t + 512] = (__bf16)h2;
    __syncthreads();
    rs[t] = h0 * h0 + h1 * h1 + h2 * h2;
    __syncthreads();
    for (int off = 128; off > 0; off >>= 1) {
        if (t < off) rs[t] += rs[t + off];
        __syncthreads();
    }
    if (t == 0) rowsq[row] = rs[0];
}

// ---------------------------------------------------------------------------
// Generic WMMA GEMM: C[4096,768] = A[4096,768](bf16) x Bt[768,768](bf16,[N][K])
// Tile 128M x 64N, K-chunk 64, 8 waves; tiles staged by the Tensor Data Mover
// into double-buffered LDS, DMA for chunk k+1 overlapped with WMMA on chunk k.
// ---------------------------------------------------------------------------
enum { EPI_Q = 0, EPI_K, EPI_V, EPI_O, EPI_RBF, EPI_FF };

template<int EPI>
__global__ __launch_bounds__(256)
void k_gemm(const __bf16* __restrict__ A, const __bf16* __restrict__ Bt,
            const float* resid, const float* bias,
            const float* __restrict__ rowsq, const float* __restrict__ colsq,
            float* outf, __bf16* __restrict__ outb) {
    __shared__ __align__(16) __bf16 sA[2][TM * LDA];
    __shared__ __align__(16) __bf16 sB[2][TN * LDB];
    int t    = threadIdx.x;
    int w    = t >> 5;
    int lane = t & 31;
    int m0   = blockIdx.x * TM;
    int n0   = blockIdx.y * TN;

    f32x8 acc[4] = {};

    // TDM prologue: stage K-chunk 0 into buffer 0 (wave 0 drives the DMA).
    if (w == 0) {
        tdm_load_tile(A  + (size_t)m0 * DQ, (unsigned)(uintptr_t)&sA[0][0], ROWS, TM);
        tdm_load_tile(Bt + (size_t)n0 * DQ, (unsigned)(uintptr_t)&sB[0][0], DQ,   TN);
    }

    int cur = 0;
    for (int kb = 0; kb < DQ; kb += TK) {
        if (w == 0) {
            if (kb + TK < DQ) {
                // issue next chunk into the other buffer, then wait for the
                // current chunk (TENSORcnt is in-order: <=2 -> oldest 2 done)
                tdm_load_tile(A  + (size_t)m0 * DQ + kb + TK,
                              (unsigned)(uintptr_t)&sA[cur ^ 1][0], ROWS, TM);
                tdm_load_tile(Bt + (size_t)n0 * DQ + kb + TK,
                              (unsigned)(uintptr_t)&sB[cur ^ 1][0], DQ, TN);
                __builtin_amdgcn_s_wait_tensorcnt(2);
            } else {
                __builtin_amdgcn_s_wait_tensorcnt(0);
            }
        }
        __syncthreads();

        const __bf16* aw = &sA[cur][(w * 16) * LDA];
        const __bf16* bw = &sB[cur][0];
        #pragma unroll
        for (int kc = 0; kc < TK; kc += 32) {
            bf16x16 af = frag_a(aw + kc, LDA, lane);
            #pragma unroll
            for (int nt = 0; nt < 4; ++nt) {
                bf16x16 bfr = frag_b(bw + (nt * 16) * LDB + kc, LDB, lane);
                acc[nt] = wmma_bf16(af, bfr, acc[nt]);
            }
        }
        __syncthreads();   // all waves done reading `cur` before DMA reuses it
        cur ^= 1;
    }

    // epilogue -- C layout: row = r + 8*half, col = lane&15
    int half  = lane >> 4;
    int mbase = m0 + w * 16 + half * 8;
    #pragma unroll
    for (int nt = 0; nt < 4; ++nt) {
        int n_g = n0 + nt * 16 + (lane & 15);
        #pragma unroll
        for (int r = 0; r < 8; ++r) {
            int   m_g = mbase + r;
            float c   = acc[nt][r];
            if (EPI == EPI_Q || EPI == EPI_K) {
                int b  = m_g >> 10, l = m_g & 1023;
                int h  = n_g >> 6,  d = n_g & 63;
                float v = (EPI == EPI_Q) ? c * 0.125f : c;   // 1/sqrt(hd)
                outb[(((size_t)(b * HQ + h)) * LQ + l) * HDQ + d] = (__bf16)v;
            } else if (EPI == EPI_V) {
                int b = m_g >> 10, l = m_g & 1023;
                int h = n_g >> 6,  d = n_g & 63;
                outb[(((size_t)(b * HQ + h)) * HDQ + d) * LQ + l] = (__bf16)c;
            } else if (EPI == EPI_O || EPI == EPI_FF) {
                size_t idx = (size_t)m_g * DQ + n_g;
                outf[idx] = resid[idx] + c + bias[n_g];
            } else { // EPI_RBF
                float sq = rowsq[m_g] + colsq[n_g] - 2.0f * c;
                sq = fmaxf(sq, 0.0f);
                outb[(size_t)m_g * DQ + n_g] = (__bf16)__expf(-0.5f * sq);
            }
        }
    }
}

// ---------------------------------------------------------------------------
// Flash attention (causal).  Grid (L/64, B*H), 128 threads = 4 waves,
// each wave owns 16 query rows; online softmax in registers via shfl_xor.
// q,k: [B*H][L][64] bf16 ; vt: [B*H][64][L] bf16 ; ctx out: [4096][768] bf16
// ---------------------------------------------------------------------------
__global__ __launch_bounds__(128)
void k_attn(const __bf16* __restrict__ q, const __bf16* __restrict__ k,
            const __bf16* __restrict__ vt, __bf16* __restrict__ ctx) {
    __shared__ __bf16 sP[4][16 * 32];
    int bh   = blockIdx.y;
    int b    = bh / HQ, h = bh - b * HQ;
    int t    = threadIdx.x, w = t >> 5, lane = t & 31;
    int half = lane >> 4, nl = lane & 15;
    int qw   = blockIdx.x * 64 + w * 16;
    __bf16* sp = sP[w];

    const __bf16* qp  = q  + ((size_t)bh * LQ + qw) * HDQ;
    const __bf16* kp0 = k  + (size_t)bh * LQ * HDQ;
    const __bf16* vtb = vt + (size_t)bh * HDQ * LQ;

    bf16x16 qf0 = frag_a(qp + 0,  HDQ, lane);
    bf16x16 qf1 = frag_a(qp + 32, HDQ, lane);

    f32x8 cacc[4] = {};
    float mrun[8], lrun[8];
    #pragma unroll
    for (int r = 0; r < 8; ++r) { mrun[r] = -1e30f; lrun[r] = 0.f; }

    const int nkb = 2 * blockIdx.x + 2;   // key blocks of 32, covers causal span
    for (int j = 0; j < nkb; ++j) {
        int kb0 = j * 32;
        // ---- S = Q K^T  (two 16-key subtiles, k=64 in two WMMA steps) ----
        f32x8 s[2];
        #pragma unroll
        for (int st = 0; st < 2; ++st) {
            const __bf16* kp = kp0 + (size_t)(kb0 + st * 16) * HDQ;
            bf16x16 kf0 = frag_b(kp + 0,  HDQ, lane);
            bf16x16 kf1 = frag_b(kp + 32, HDQ, lane);
            f32x8 c = {};
            c = wmma_bf16(qf0, kf0, c);
            c = wmma_bf16(qf1, kf1, c);
            s[st] = c;
        }
        // ---- causal mask + online softmax (rows live at (r, lane-half)) --
        #pragma unroll
        for (int r = 0; r < 8; ++r) {
            int qrow = qw + half * 8 + r;
            float s0 = s[0][r]; if (kb0 + nl      > qrow) s0 = -1e30f;
            float s1 = s[1][r]; if (kb0 + 16 + nl > qrow) s1 = -1e30f;
            float rm = fmaxf(s0, s1);
            rm = fmaxf(rm, __shfl_xor(rm, 1));
            rm = fmaxf(rm, __shfl_xor(rm, 2));
            rm = fmaxf(rm, __shfl_xor(rm, 4));
            rm = fmaxf(rm, __shfl_xor(rm, 8));
            float mnew = fmaxf(mrun[r], rm);
            float sc   = __expf(mrun[r] - mnew);
            mrun[r] = mnew;
            float p0 = __expf(s0 - mnew);
            float p1 = __expf(s1 - mnew);
            float rs = p0 + p1;
            rs += __shfl_xor(rs, 1);
            rs += __shfl_xor(rs, 2);
            rs += __shfl_xor(rs, 4);
            rs += __shfl_xor(rs, 8);
            lrun[r] = lrun[r] * sc + rs;
            #pragma unroll
            for (int nt = 0; nt < 4; ++nt) cacc[nt][r] *= sc;
            sp[(half * 8 + r) * 32 + nl]      = (__bf16)p0;
            sp[(half * 8 + r) * 32 + 16 + nl] = (__bf16)p1;
        }
        __syncthreads();
        // ---- ctx += P @ V ----
        bf16x16 pf = frag_a(sp, 32, lane);
        #pragma unroll
        for (int nt = 0; nt < 4; ++nt) {
            bf16x16 vf = frag_b(vtb + (size_t)(nt * 16) * LQ + kb0, LQ, lane);
            cacc[nt] = wmma_bf16(pf, vf, cacc[nt]);
        }
        __syncthreads();
    }
    // ---- normalize + scatter back to [B,L,D] bf16 ----
    size_t row0 = (size_t)b * LQ + qw + half * 8;
    #pragma unroll
    for (int nt = 0; nt < 4; ++nt) {
        int col = h * HDQ + nt * 16 + nl;
        #pragma unroll
        for (int r = 0; r < 8; ++r) {
            float v = cacc[nt][r] / lrun[r];
            ctx[(row0 + r) * DQ + col] = (__bf16)v;
        }
    }
}

// ---------------------------------------------------------------------------
// Launch
// ---------------------------------------------------------------------------
extern "C" void kernel_launch(void* const* d_in, const int* in_sizes, int n_in,
                              void* d_out, int out_size, void* d_ws, size_t ws_size,
                              hipStream_t stream) {
    (void)in_sizes; (void)n_in; (void)out_size; (void)ws_size;
    const float* x   = (const float*)d_in[0];
    const float* Wq  = (const float*)d_in[1];
    const float* Wk  = (const float*)d_in[2];
    const float* Wv  = (const float*)d_in[3];
    const float* Wo  = (const float*)d_in[4];
    const float* bo  = (const float*)d_in[5];
    const float* s1  = (const float*)d_in[6];
    const float* sh1 = (const float*)d_in[7];
    const float* s2  = (const float*)d_in[8];
    const float* sh2 = (const float*)d_in[9];
    const float* cen = (const float*)d_in[10];
    const float* ls  = (const float*)d_in[11];
    const float* Wf  = (const float*)d_in[12];
    const float* bfv = (const float*)d_in[13];
    float* out = (float*)d_out;

    char*  wsb = (char*)d_ws;
    size_t off = 0;
    auto take = [&](size_t bytes) -> void* {
        void* p = wsb + off;
        off += (bytes + 255) & ~(size_t)255;
        return p;
    };
    __bf16* h1   = (__bf16*)take((size_t)ROWS * DQ * 2);
    __bf16* wtq  = (__bf16*)take((size_t)DQ * DQ * 2);
    __bf16* wtk  = (__bf16*)take((size_t)DQ * DQ * 2);
    __bf16* wtv  = (__bf16*)take((size_t)DQ * DQ * 2);
    __bf16* wto  = (__bf16*)take((size_t)DQ * DQ * 2);
    __bf16* wtf  = (__bf16*)take((size_t)DQ * DQ * 2);
    __bf16* cfb  = (__bf16*)take((size_t)DQ * DQ * 2);
    float*  csum = (float*)take((size_t)DQ * 4);
    __bf16* qb   = (__bf16*)take((size_t)ROWS * DQ * 2);
    __bf16* kb   = (__bf16*)take((size_t)ROWS * DQ * 2);
    __bf16* vtb  = (__bf16*)take((size_t)ROWS * DQ * 2);
    __bf16* ctxb = (__bf16*)take((size_t)ROWS * DQ * 2);
    float*  rowsq= (float*)take((size_t)ROWS * 4);
    __bf16* hfb  = qb;   // reuse: q/k dead after attention
    __bf16* kmat = kb;

    dim3 tb(32, 8), tg(24, 24);
    k_transpose_bf16<<<tg, tb, 0, stream>>>(Wq, wtq);
    k_transpose_bf16<<<tg, tb, 0, stream>>>(Wk, wtk);
    k_transpose_bf16<<<tg, tb, 0, stream>>>(Wv, wtv);
    k_transpose_bf16<<<tg, tb, 0, stream>>>(Wo, wto);
    k_transpose_bf16<<<tg, tb, 0, stream>>>(Wf, wtf);
    k_centers<<<DQ, 256, 0, stream>>>(cen, ls, cfb, csum);
    k_ln1<<<ROWS, 256, 0, stream>>>(x, s1, sh1, h1);

    dim3 gg(ROWS / TM, DQ / TN);
    k_gemm<EPI_Q  ><<<gg, 256, 0, stream>>>(h1, wtq, nullptr, nullptr, nullptr, nullptr, nullptr, qb);
    k_gemm<EPI_K  ><<<gg, 256, 0, stream>>>(h1, wtk, nullptr, nullptr, nullptr, nullptr, nullptr, kb);
    k_gemm<EPI_V  ><<<gg, 256, 0, stream>>>(h1, wtv, nullptr, nullptr, nullptr, nullptr, nullptr, vtb);

    k_attn<<<dim3(LQ / 64, BQ * HQ), 128, 0, stream>>>(qb, kb, vtb, ctxb);

    k_gemm<EPI_O  ><<<gg, 256, 0, stream>>>(ctxb, wto, x, bo, nullptr, nullptr, out, nullptr);
    k_ln2<<<ROWS, 256, 0, stream>>>(out, s2, sh2, ls, hfb, rowsq);
    k_gemm<EPI_RBF><<<gg, 256, 0, stream>>>(hfb, cfb, nullptr, nullptr, rowsq, csum, nullptr, kmat);
    k_gemm<EPI_FF ><<<gg, 256, 0, stream>>>(kmat, wtf, out, bfv, nullptr, nullptr, out, nullptr);
}
